// FiberConv_82617990906072
// MI455X (gfx1250) — matrix operand order repaired and checked by reading, not compiled, not run on hardware
//
#include <hip/hip_runtime.h>

// FiberConv as one GEMM: out[400000, 384] = X[400000, 384] @ Kmat[384, 384]
// Kmat[(s,i),(t,o)] = sum_a edge_attr[t,s,a] * W[o*32+i, a]
// bf16 hi/lo split (3x V_WMMA_F32_16X16X32_BF16) for fp32-class accuracy.
// NT stores protect L2 so the 4 column-group passes re-read X from L2, not HBM.

#define S_DIM   12
#define ATTR    3
#define IN_DIM  32
#define OUT_DIM 32
#define BATCH   400000
#define KTOT    384           // S*IN
#define NTOT    384           // T*OUT
#define NCOLS   96            // N columns per workgroup (4 column groups)
#define NTILES  (NCOLS / 16)  // 6
#define KPAD    392           // 384 + 8 bf16 pad -> kills LDS bank conflicts
#define MBLOCK  128           // 8 waves * 16 rows
#define MGRID   625
#define MBLOCKS (BATCH / MBLOCK)  // 3125 (exact)

typedef __attribute__((ext_vector_type(16))) __bf16 v16bf;
typedef __attribute__((ext_vector_type(8)))  __bf16 v8bf;
typedef __attribute__((ext_vector_type(8)))  float  v8f;

__device__ __forceinline__ void split_a(const float* __restrict__ ap,
                                        v16bf& ah, v16bf& al)
{
    const float4 f0 = *(const float4*)(ap);
    const float4 f1 = *(const float4*)(ap + 4);
    const float4 f2 = *(const float4*)(ap + 16);
    const float4 f3 = *(const float4*)(ap + 20);
    const float fv[16] = { f0.x, f0.y, f0.z, f0.w,  f1.x, f1.y, f1.z, f1.w,
                           f2.x, f2.y, f2.z, f2.w,  f3.x, f3.y, f3.z, f3.w };
    #pragma unroll
    for (int j = 0; j < 16; ++j) {
        const __bf16 h = (__bf16)fv[j];
        ah[j] = h;
        al[j] = (__bf16)(fv[j] - (float)h);
    }
}

__global__ __launch_bounds__(256)
void fiberconv_wmma(const float* __restrict__ x,
                    const float* __restrict__ edge_attr,
                    const float* __restrict__ W,
                    const float* __restrict__ bias,
                    float* __restrict__ out)
{
    // B slice, transposed: Bt[n][k], hi/lo bf16.  2 * 96 * 392 * 2B = 147 KB LDS.
    __shared__ __bf16 Bh[NCOLS][KPAD];
    __shared__ __bf16 Bl[NCOLS][KPAD];

    const int tid   = threadIdx.x;
    const int lane  = tid & 31;
    const int wave  = tid >> 5;
    const int nbase = blockIdx.x * NCOLS;

    // ---- build kernel-matrix slice in LDS (tiny: 3 FMAs per element) ----
    for (int idx = tid; idx < NCOLS * KTOT; idx += 256) {
        const int n  = idx / KTOT;
        const int k  = idx - n * KTOT;
        const int gn = nbase + n;
        const int t = gn >> 5, o = gn & 31;   // col = t*32 + o
        const int s = k  >> 5, i = k  & 31;   // row = s*32 + i
        const float* ea = edge_attr + (t * S_DIM + s) * ATTR;
        const float* wr = W + (o * IN_DIM + i) * ATTR;
        const float v = ea[0] * wr[0] + ea[1] * wr[1] + ea[2] * wr[2];
        const __bf16 h = (__bf16)v;
        Bh[n][k] = h;
        Bl[n][k] = (__bf16)(v - (float)h);
    }
    __syncthreads();

    const int lhalf  = lane >> 4;   // 0 | 1
    const int l15    = lane & 15;
    const int ka_off = lhalf * 8;   // A lane-half K offset (16-bit A 16x32 layout)
    const int kb_off = lhalf * 8;   // B lane-half K offset (16-bit B 32x16 layout)

    for (int blk = blockIdx.y; blk < MBLOCKS; blk += MGRID) {
        const int m0 = blk * MBLOCK + wave * 16;
        const float* arow = x + (size_t)(m0 + l15) * KTOT;

        v8f acc[NTILES] = {};

        #pragma unroll 2
        for (int k0 = 0; k0 < KTOT; k0 += 32) {
            const float* ap = arow + k0 + ka_off;
            // Pull the 128B lines two K-chunks ahead (k0+64) toward the WGP while WMMAs run.
            // Speculative prefetch: past-the-end addresses are silently dropped.
            __builtin_prefetch(ap + 64, 0, 3);

            v16bf ah, al;
            split_a(ap, ah, al);

            #pragma unroll
            for (int t = 0; t < NTILES; ++t) {
                const __bf16* bp = &Bh[t * 16 + l15][k0 + kb_off];
                const __bf16* lp = &Bl[t * 16 + l15][k0 + kb_off];
                const v8bf bh0 = *(const v8bf*)bp;
                const v8bf bh1 = *(const v8bf*)(bp + 16);
                const v8bf bl0 = *(const v8bf*)lp;
                const v8bf bl1 = *(const v8bf*)(lp + 16);
                const v16bf bh = __builtin_shufflevector(
                    bh0, bh1, 0,1,2,3,4,5,6,7,8,9,10,11,12,13,14,15);
                const v16bf bl = __builtin_shufflevector(
                    bl0, bl1, 0,1,2,3,4,5,6,7,8,9,10,11,12,13,14,15);

                acc[t] = __builtin_amdgcn_wmma_f32_16x16x32_bf16(
                    false, ah, false, bh, (short)0, acc[t], false, false);
                acc[t] = __builtin_amdgcn_wmma_f32_16x16x32_bf16(
                    false, ah, false, bl, (short)0, acc[t], false, false);
                acc[t] = __builtin_amdgcn_wmma_f32_16x16x32_bf16(
                    false, al, false, bh, (short)0, acc[t], false, false);
            }
        }

        // ---- store C: VGPR v, lanes 0-15 -> M=v, lanes 16-31 -> M=v+8 ----
        // Non-temporal: out is write-once; keep it from evicting X's L2 lines
        // (the other 3 column groups re-read X from L2).
        const int rbase = m0 + lhalf * 8;
        #pragma unroll
        for (int t = 0; t < NTILES; ++t) {
            const int gcol = nbase + t * 16 + l15;
            const float bv = bias[gcol & (OUT_DIM - 1)];
            float* op = out + (size_t)rbase * NTOT + gcol;
            #pragma unroll
            for (int v = 0; v < 8; ++v)
                __builtin_nontemporal_store(acc[t][v] + bv, op + (size_t)v * NTOT);
        }
    }
}

extern "C" void kernel_launch(void* const* d_in, const int* in_sizes, int n_in,
                              void* d_out, int out_size, void* d_ws, size_t ws_size,
                              hipStream_t stream) {
    const float* x  = (const float*)d_in[0];
    const float* ea = (const float*)d_in[1];
    const float* W  = (const float*)d_in[2];
    const float* b  = (const float*)d_in[3];
    float* out = (float*)d_out;
    (void)in_sizes; (void)n_in; (void)out_size; (void)d_ws; (void)ws_size;

    dim3 grid(NTOT / NCOLS, MGRID, 1);   // (4 column groups, 625 M groups)
    fiberconv_wmma<<<grid, dim3(256, 1, 1), 0, stream>>>(x, ea, W, b, out);
}
